// DoubleTrajGRUBlock_17343077941767
// MI455X (gfx1250) — matrix operand order repaired and checked by reading, not compiled, not run on hardware
//
#include <hip/hip_runtime.h>
#include <hip/hip_bf16.h>

// DoubleTrajGRUBlock for MI455X (gfx1250).
// All convolutions run as implicit-GEMM bf16 WMMA (v_wmma_f32_16x16x32_bf16,
// f32 accumulate).  A-tiles are cooperatively staged in LDS (double-buffered)
// so the 8 waves of a block share one gather; fragments are then read as
// 16-byte ds_load_b128s.

typedef __attribute__((ext_vector_type(16))) __bf16 v16bf;
typedef __attribute__((ext_vector_type(8)))  float  v8f;

#define NEG_SLOPE 0.2f

// Problem geometry is fixed: 8 x C x 128 x 128 tensors.
constexpr int G_H  = 128;
constexpr int G_W  = 128;
constexpr int G_HW = G_H * G_W;     // 16384

__device__ __forceinline__ unsigned short f2bf(float f) {
  union { float f; unsigned u; } v; v.f = f;
  unsigned u = v.u;
  u += 0x7FFFu + ((u >> 16) & 1u);       // round to nearest even
  return (unsigned short)(u >> 16);
}

// ---------------------------------------------------------------------------
// Implicit-GEMM convolution with WMMA + LDS-staged A tiles.
//   in_bf : NCHW bf16 activations, Ctot channels; reads channels
//           [ch_off, ch_off+Cin).
//   Bmat  : repacked weights, [Npad][K] bf16, K = Cin*KW*KW (mult of 32),
//           row-major in K (each lane's 16-wide K slice is contiguous).
//   out   : NCHW f32, Cout channels; outbf: optional bf16 duplicate.
// Block = 8 waves covering MW M-tiles x (NW*NT) N-tiles of one M-slab.
//   MW * NW == 8; wave w -> msub = w/NW, nsub = w%NW (NT tiles each).
// ---------------------------------------------------------------------------
template<int KW, int MW, int NT>
__global__ __launch_bounds__(256)
void conv_wmma_kernel(
    const unsigned short* __restrict__ inbf, int ch_off, int Cin, int Ctot,
    const unsigned short* __restrict__ Bmat,
    const float* __restrict__ bias0, const float* __restrict__ bias1,
    float* __restrict__ out, unsigned short* __restrict__ outbf,
    int Cout, int pad, int leaky)
{
  constexpr int KK     = KW * KW;
  constexpr int NW     = 8 / MW;            // waves along N
  constexpr int AROWS  = MW * 16;           // M rows staged per block
  constexpr int LOG2AR = (MW == 1) ? 4 : (MW == 2) ? 5 : (MW == 4) ? 6 : 7;
  constexpr int ASTR   = 40;                // row stride in ushorts (80B: 16B-aligned)

  __shared__ __align__(16) unsigned short ldsA[2][AROWS * ASTR];

  const int K    = Cin * KK;
  const int tid  = threadIdx.x;
  const int lane = tid & 31;
  const int w    = tid >> 5;                // wave in block: 0..7
  const int msub = w / NW;
  const int nsub = w % NW;
  const int half = lane >> 4;               // 0 or 1
  const int l16  = lane & 15;

  const int mbase0 = blockIdx.x * AROWS;    // first pixel row of this block's slab
  const int mbase  = mbase0 + msub * 16;    // this wave's M tile
  const int nbase  = nsub * (16 * NT);      // this wave's first N tile

  // Base of this lane's B rows; tile j adds j*16 rows.
  const unsigned short* Bbase = Bmat + (size_t)(nbase + l16) * K;

  // ---- Hoist per-staging-element pixel decomposition (kb-invariant).
  // Element e = tid + i*256:  m_loc = e & (AROWS-1), k_loc = e >> LOG2AR.
  int st_kloc[MW * 2], st_y[MW * 2], st_x[MW * 2], st_mloc[MW * 2];
  size_t st_base[MW * 2];
#pragma unroll
  for (int i = 0; i < MW * 2; ++i) {
    const int e     = tid + i * 256;
    const int m_loc = e & (AROWS - 1);
    const int k_loc = e >> LOG2AR;
    const int m     = mbase0 + m_loc;
    const int nbat  = m >> 14;              // / G_HW
    const int rA    = m & (G_HW - 1);
    st_mloc[i] = m_loc;
    st_kloc[i] = k_loc;
    st_y[i]    = (rA >> 7);
    st_x[i]    = (rA & (G_W - 1));
    st_base[i] = (size_t)(nbat * Ctot + ch_off) * G_HW;
  }

  v8f acc[NT];
#pragma unroll
  for (int j = 0; j < NT; ++j) acc[j] = (v8f){};

  int parity = 0;
  for (int kb = 0; kb < K; kb += 32) {
    // ---- Cooperative stage of A[AROWS][32] slice into LDS.
#pragma unroll
    for (int i = 0; i < MW * 2; ++i) {
      const int k    = kb + st_kloc[i];
      const int ci   = k / KK;              // constexpr KK -> mul/shift
      const int kidx = k - ci * KK;
      const int ky   = kidx / KW;
      const int kx   = kidx - ky * KW;
      const int iy   = st_y[i] + ky - pad;
      const int ix   = st_x[i] + kx - pad;
      unsigned short v = 0;
      if ((unsigned)iy < (unsigned)G_H && (unsigned)ix < (unsigned)G_W)
        v = inbf[st_base[i] + ((size_t)ci << 14) + (iy << 7) + ix];
      ldsA[parity][st_mloc[i] * ASTR + st_kloc[i]] = v;
    }
    __syncthreads();

    // ---- A fragment: slots 0..15 = K kb+half*16..+16 (contiguous, matches B)
    //      -> two 16B LDS loads.
    union { v16bf v; uint4 q[2]; } afrag;
    const unsigned short* ar = &ldsA[parity][(msub * 16 + l16) * ASTR];
    afrag.q[0] = *(const uint4*)(ar + half * 16);
    afrag.q[1] = *(const uint4*)(ar + half * 16 + 8);

    // ---- NT B fragments feed NT WMMAs off the same A fragment.
#pragma unroll
    for (int j = 0; j < NT; ++j) {
      union { v16bf v; uint4 q[2]; } bfrag;
      const unsigned short* bp = Bbase + (size_t)j * 16 * K + kb + half * 16;
      bfrag.q[0] = ((const uint4*)bp)[0];
      bfrag.q[1] = ((const uint4*)bp)[1];
      if (j == 0 && kb + 32 < K)
        __builtin_prefetch(bp + 32, 0, 0);   // global_prefetch_b8
      acc[j] = __builtin_amdgcn_wmma_f32_16x16x32_bf16(
          false, afrag.v, false, bfrag.v, (short)0, acc[j], false, false);
    }
    parity ^= 1;   // double buffer: one barrier per K step, no WAR hazard
  }

  // ---- Epilogue: C/D layout — VGPR v holds M = v + 8*half, N = l16.
#pragma unroll
  for (int j = 0; j < NT; ++j) {
    const int co = nbase + j * 16 + l16;
    if (co < Cout) {
      const float bs = bias0[co] + (bias1 ? bias1[co] : 0.0f);
#pragma unroll
      for (int v = 0; v < 8; ++v) {
        const int m  = mbase + half * 8 + v;
        float val = acc[j][v] + bs;
        if (leaky) val = (val >= 0.0f) ? val : NEG_SLOPE * val;
        const int nb = m >> 14;
        const int r  = m & (G_HW - 1);
        const size_t o = (size_t)(nb * Cout + co) * G_HW + r;
        out[o] = val;
        if (outbf) outbf[o] = f2bf(val);
      }
    }
  }
}

// Cast an NCHW f32 tensor into a bf16 NCHW tensor with Ctot channels,
// placing it at channel offset ch_off (for channel concatenation).
__global__ void cast_bf16_kernel(const float* __restrict__ src,
                                 unsigned short* __restrict__ dst,
                                 int C, int ch_off, int Ctot, int Batch)
{
  const int i = blockIdx.x * blockDim.x + threadIdx.x;
  const int n = Batch * C * G_HW;
  if (i >= n) return;
  const int p = i & (G_HW - 1);
  const int t = i >> 14;
  const int c = t % C;
  const int b = t / C;
  dst[(size_t)(b * Ctot + ch_off + c) * G_HW + p] = f2bf(src[i]);
}

__global__ void zero_u16_kernel(unsigned short* __restrict__ p, int n)
{
  const int i = blockIdx.x * blockDim.x + threadIdx.x;
  if (i < n) p[i] = 0;
}

// Repack conv weight (O, I, k, k) f32 -> bf16 B-matrix [Npad][Ktot],
// rows = output channel, K index = (ci_off + ci)*kk + kidx (matches the
// A-gather order above; ci_off supports channel-concat fusion).
__global__ void repack_w_kernel(const float* __restrict__ w,
                                unsigned short* __restrict__ Bmat,
                                int O, int I, int kk, int Ktot, int ci_off)
{
  const int i = blockIdx.x * blockDim.x + threadIdx.x;
  const int n = O * I * kk;
  if (i >= n) return;
  const int kidx = i % kk;
  const int t    = i / kk;
  const int ci   = t % I;
  const int o    = t / I;
  const int k    = (ci_off + ci) * kk + kidx;
  Bmat[(size_t)o * Ktot + k] = f2bf(w[i]);
}

// Bilinear warp: for each (b, l, y, x) sample h at (x - fx, y - fy) across all
// F channels; zero outside the image.  Output bf16 NCHW with C = L*F (this is
// the A-matrix of the 1x1 "ret" conv).
__global__ void warp_kernel(const float* __restrict__ h,
                            const float* __restrict__ flows,
                            unsigned short* __restrict__ warped,
                            int Batch, int F, int Lnum)
{
  const int i = blockIdx.x * blockDim.x + threadIdx.x;
  const int n = Batch * Lnum * G_HW;
  if (i >= n) return;
  const int p = i & (G_HW - 1);
  const int t = i >> 14;
  const int l = t % Lnum;
  const int b = t / Lnum;
  const int y = p >> 7;
  const int x = p & (G_W - 1);

  const float fx = flows[(size_t)(b * (2 * Lnum) + 2 * l + 0) * G_HW + p];
  const float fy = flows[(size_t)(b * (2 * Lnum) + 2 * l + 1) * G_HW + p];
  const float px = (float)x - fx;
  const float py = (float)y - fy;
  const float x0f = floorf(px), y0f = floorf(py);
  const float wx = px - x0f, wy = py - y0f;
  const int x0 = (int)x0f, y0 = (int)y0f;
  const int x1 = x0 + 1,   y1 = y0 + 1;
  const bool vx0 = (x0 >= 0) && (x0 < G_W);
  const bool vx1 = (x1 >= 0) && (x1 < G_W);
  const bool vy0 = (y0 >= 0) && (y0 < G_H);
  const bool vy1 = (y1 >= 0) && (y1 < G_H);
  const int xc0 = min(max(x0, 0), G_W - 1), xc1 = min(max(x1, 0), G_W - 1);
  const int yc0 = min(max(y0, 0), G_H - 1), yc1 = min(max(y1, 0), G_H - 1);
  const float w00 = (vx0 && vy0) ? (1.f - wx) * (1.f - wy) : 0.f;
  const float w10 = (vx1 && vy0) ? wx * (1.f - wy) : 0.f;
  const float w01 = (vx0 && vy1) ? (1.f - wx) * wy : 0.f;
  const float w11 = (vx1 && vy1) ? wx * wy : 0.f;
  const int o00 = (yc0 << 7) + xc0, o10 = (yc0 << 7) + xc1;
  const int o01 = (yc1 << 7) + xc0, o11 = (yc1 << 7) + xc1;

  const float* hb = h + (size_t)b * F * G_HW;
  unsigned short* ob = warped + (size_t)(b * (Lnum * F) + l * F) * G_HW + p;
  for (int c = 0; c < F; ++c) {
    const float* hc = hb + ((size_t)c << 14);
    const float v = w00 * hc[o00] + w10 * hc[o10] + w01 * hc[o01] + w11 * hc[o11];
    ob[(size_t)c << 14] = f2bf(v);
  }
}

// GRU gate fusion: r = sig(i0+h0), u = sig(i1+h1),
// mem = leaky(i2 + r*h2), h' = u*h + (1-u)*mem.
// Optionally also writes h' in bf16 into a concat buffer for the next step.
__global__ void gate_kernel(const float* __restrict__ i2h,
                            const float* __restrict__ h2h,
                            const float* __restrict__ hprev,
                            float* __restrict__ hout,
                            unsigned short* __restrict__ houtbf,
                            int ch_off, int Ctot, int Batch, int F)
{
  const int i = blockIdx.x * blockDim.x + threadIdx.x;
  const int n = Batch * F * G_HW;
  if (i >= n) return;
  const int p = i & (G_HW - 1);
  const int t = i >> 14;
  const int c = t % F;
  const int b = t / F;
  const size_t base = (size_t)(b * 3 * F + c) * G_HW + p;
  const size_t sF   = (size_t)F * G_HW;
  const float i0 = i2h[base],         h0 = h2h[base];
  const float i1 = i2h[base + sF],    h1 = h2h[base + sF];
  const float i2v = i2h[base + 2*sF], h2v = h2h[base + 2*sF];
  const float r = 1.f / (1.f + expf(-(i0 + h0)));
  const float u = 1.f / (1.f + expf(-(i1 + h1)));
  float mem = i2v + r * h2v;
  mem = (mem >= 0.f) ? mem : NEG_SLOPE * mem;
  const float hp = hprev[(size_t)(b * F + c) * G_HW + p];
  const float val = u * hp + (1.f - u) * mem;
  hout[(size_t)(b * F + c) * G_HW + p] = val;
  if (houtbf)
    houtbf[(size_t)(b * Ctot + ch_off + c) * G_HW + p] = f2bf(val);
}

// ---------------------------------------------------------------------------
template<int KW, int MW, int NT>
static void launch_conv(const unsigned short* inbf, int ch_off, int Cin, int Ctot,
                        const unsigned short* Bm, const float* b0, const float* b1,
                        float* out, unsigned short* outbf, int Cout,
                        int Bn, int pad, int leaky, hipStream_t stream)
{
  const int blocks = (Bn * G_HW / 16) / MW;  // one M-slab (MW tiles) per block
  conv_wmma_kernel<KW, MW, NT><<<blocks, 256, 0, stream>>>(
      inbf, ch_off, Cin, Ctot, Bm, b0, b1, out, outbf, Cout, pad, leaky);
}

extern "C" void kernel_launch(void* const* d_in, const int* in_sizes, int n_in,
                              void* d_out, int out_size, void* d_ws, size_t ws_size,
                              hipStream_t stream)
{
  (void)in_sizes; (void)n_in; (void)out_size; (void)ws_size;
  const int Bn = 8, F = 128, Lnum = 5, FH = 32;

  const float* x      = (const float*)d_in[0];
  const float* hprev1 = (const float*)d_in[1];
  const float* hprev2 = (const float*)d_in[2];
  const float* W1[10], *W2[10];
  for (int i = 0; i < 10; ++i) { W1[i] = (const float*)d_in[3 + i]; W2[i] = (const float*)d_in[13 + i]; }
  // W[0]=i2h_w (3F,Cin,3,3) W[1]=i2h_b W[2]=i2f_w (32,Cin,5,5) W[3]=i2f_b
  // W[4]=h2f_w (32,F,5,5)   W[5]=h2f_b W[6]=flow_w (10,32,5,5) W[7]=flow_b
  // W[8]=ret_w (3F,5F,1,1)  W[9]=ret_b

  char* wsp = (char*)d_ws;
  size_t off = 0;
  auto alloc = [&](size_t bytes) -> void* {
    void* p = wsp + off;
    off += (bytes + 255) & ~(size_t)255;
    return p;
  };

  unsigned short* xh1    = (unsigned short*)alloc((size_t)Bn * 192 * G_HW * 2);
  unsigned short* xh2    = (unsigned short*)alloc((size_t)Bn * 256 * G_HW * 2);
  float*          f1     = (float*)         alloc((size_t)Bn * FH * G_HW * 4);
  unsigned short* f1b    = (unsigned short*)alloc((size_t)Bn * FH * G_HW * 2);
  float*          flows  = (float*)         alloc((size_t)Bn * 2 * Lnum * G_HW * 4);
  unsigned short* warped = (unsigned short*)alloc((size_t)Bn * Lnum * F * G_HW * 2);
  float*          h2h    = (float*)         alloc((size_t)Bn * 3 * F * G_HW * 4);
  float*          i2h    = (float*)         alloc((size_t)Bn * 3 * F * G_HW * 4);
  float*          h1     = (float*)         alloc((size_t)Bn * F * G_HW * 4);
  unsigned short* Bfw    = (unsigned short*)alloc((size_t)32 * 6400 * 2);
  unsigned short* Bflow  = (unsigned short*)alloc((size_t)16 * 800 * 2);
  unsigned short* Bret   = (unsigned short*)alloc((size_t)384 * 640 * 2);
  unsigned short* Bi2h   = (unsigned short*)alloc((size_t)384 * 1152 * 2);

  auto blocks1d = [](int n) { return dim3((unsigned)((n + 255) / 256)); };

  auto step = [&](const unsigned short* xh, int Cin, int Ctot, const float* hp,
                  const float* const* W,
                  float* hout, unsigned short* houtbf, int ho_off, int ho_Ctot) {
    // f1 = leaky(conv5x5([x;h], [i2f;h2f]) + i2f_b + h2f_b)   (fused concat conv)
    const int Kfw = Ctot * 25;
    repack_w_kernel<<<blocks1d(32 * Cin * 25), 256, 0, stream>>>(W[2], Bfw, 32, Cin, 25, Kfw, 0);
    repack_w_kernel<<<blocks1d(32 * F * 25),   256, 0, stream>>>(W[4], Bfw, 32, F,   25, Kfw, Cin);
    launch_conv<5, 4, 1>(xh, 0, Ctot, Ctot, Bfw, W[3], W[5], f1, f1b, 32,
                         Bn, 2, /*leaky=*/1, stream);          // Npad=32: NW=2

    // flows = conv5x5(f1, flow_w) + flow_b        (Cout=10, N padded to 16)
    zero_u16_kernel<<<blocks1d(16 * 800), 256, 0, stream>>>(Bflow, 16 * 800);
    repack_w_kernel<<<blocks1d(10 * 32 * 25), 256, 0, stream>>>(W[6], Bflow, 10, 32, 25, 800, 0);
    launch_conv<5, 8, 1>(f1b, 0, 32, 32, Bflow, W[7], nullptr, flows, nullptr, 10,
                         Bn, 2, 0, stream);                    // Npad=16: NW=1

    // warped[l] = bilinear(h, x - flow_x, y - flow_y)   -> bf16, C = L*F
    warp_kernel<<<blocks1d(Bn * Lnum * G_HW), 256, 0, stream>>>(hp, flows, warped, Bn, F, Lnum);

    // h2h = conv1x1(warped, ret_w) + ret_b        (GEMM M=131072 N=384 K=640)
    repack_w_kernel<<<blocks1d(384 * 640), 256, 0, stream>>>(W[8], Bret, 384, Lnum * F, 1, 640, 0);
    launch_conv<1, 1, 3>(warped, 0, Lnum * F, Lnum * F, Bret, W[9], nullptr, h2h, nullptr,
                         384, Bn, 0, 0, stream);               // Npad=384: NW=8, NT=3

    // i2h = conv3x3(x, i2h_w) + i2h_b             (reads x-part of concat buf)
    const int Ki = Cin * 9;
    repack_w_kernel<<<blocks1d(384 * Cin * 9), 256, 0, stream>>>(W[0], Bi2h, 384, Cin, 9, Ki, 0);
    launch_conv<3, 1, 3>(xh, 0, Cin, Ctot, Bi2h, W[1], nullptr, i2h, nullptr,
                         384, Bn, 1, 0, stream);               // Npad=384: NW=8, NT=3

    // gates -> h'
    gate_kernel<<<blocks1d(Bn * F * G_HW), 256, 0, stream>>>(
        i2h, h2h, hp, hout, houtbf, ho_off, ho_Ctot, Bn, F);
  };

  // Prep: bf16 concat buffers.  xh1 = [x(64) ; h_prev1(128)], xh2 gets
  // h_prev2 in channels [128,256); step 1's gate writes h1 into [0,128).
  cast_bf16_kernel<<<blocks1d(Bn * 64 * G_HW), 256, 0, stream>>>(x,      xh1, 64, 0,   192, Bn);
  cast_bf16_kernel<<<blocks1d(Bn * F * G_HW),  256, 0, stream>>>(hprev1, xh1, F,  64,  192, Bn);
  cast_bf16_kernel<<<blocks1d(Bn * F * G_HW),  256, 0, stream>>>(hprev2, xh2, F,  128, 256, Bn);

  step(xh1, 64,  192, hprev1, W1, h1,            xh2,     0, 256);
  step(xh2, 128, 256, hprev2, W2, (float*)d_out, nullptr, 0, 0);
}
